// SelfAttention_18038862643593
// MI455X (gfx1250) — compile-verified, matrix operand order checked by reading, not run on hardware
//
#include <hip/hip_runtime.h>
#include <hip/hip_bf16.h>
#include <cstdint>
#include <cstddef>

// CDNA5 (gfx1250) wave32 WMMA self-attention pipeline.
//   - v_wmma_f32_16x16x32_f16 for all matmuls (f16 in, f32 accum)
//   - global_load_async_to_lds_b128 (ASYNCcnt) tile staging, double-buffered
//   - K-step 64 (two WMMA sub-steps per barrier)

typedef __attribute__((ext_vector_type(16))) _Float16 v16h;
typedef __attribute__((ext_vector_type(8)))  float    v8f;
typedef int v4i_t __attribute__((vector_size(16)));   // b128 element type

#define D_DIM  1024
#define S_LEN  2048
#define NBATCH 4
#define KSTEP  64
#define LSTR   72   // LDS tile row stride in halves: 144B/row -> frags 16B-aligned

// ---------------------------------------------------------------------------
// Async global->LDS 16-byte copy. Tier 2: clang builtin (signature takes
// int4-vector pointers: AS1 src printed as `__device__`, AS3 dst);
// tier 1: inline asm fallback.
// ---------------------------------------------------------------------------
#if __has_builtin(__builtin_amdgcn_global_load_async_to_lds_b128) && \
    __has_builtin(__builtin_amdgcn_s_wait_asynccnt)
#define ASYNC_MODE 2
#else
#define ASYNC_MODE 1
#endif

__device__ __forceinline__ void cp16_async(const _Float16* g, _Float16* l) {
#if ASYNC_MODE == 2
  __builtin_amdgcn_global_load_async_to_lds_b128(
      (__attribute__((address_space(1))) v4i_t*)g,
      (__attribute__((address_space(3))) v4i_t*)l, 0, 0);
#else
  // LDS flat address: low 32 bits are the LDS byte offset (ISA 10.2 aperture)
  const unsigned lofs = (unsigned)(unsigned long long)l;
  asm volatile("global_load_async_to_lds_b128 %0, %1, off"
               :: "v"(lofs), "v"(g) : "memory");
#endif
}

__device__ __forceinline__ void wait_stage() {
#if ASYNC_MODE == 2
  __builtin_amdgcn_s_wait_asynccnt(0);
#else
  asm volatile("s_wait_asynccnt 0x0" ::: "memory");
#endif
  __syncthreads();   // compiler adds s_wait_dscnt for the plain-store paths
}

// ---------------------------------------------------------------------------
// Fragment gather from an LDS tile laid out [idx][k] (stride LSTR halves).
// ISA 7.12.2 (16-bit A 16x32): lane L holds m/n = L&15; halves cover
// K = [g8, g8+8) and [16+g8, 16+g8+8), g8 = (L>>4)*8 -> two b128 LDS reads.
// ---------------------------------------------------------------------------
__device__ __forceinline__ v16h ld_frag(const _Float16* lds, int idx0, int ko) {
  const int lane = threadIdx.x & 31;
  const int idx  = idx0 + (lane & 15);
  const int g8   = (lane >> 4) << 3;
  const _Float16* p = lds + idx * LSTR + ko + g8;
  union { v16h h; uint4 u[2]; } f;
  f.u[0] = *(const uint4*)(p);
  f.u[1] = *(const uint4*)(p + 16);
  return f.h;
}

// Stage one 128x64 f16 A tile and one 128x64 f16 B tile (both row-contiguous
// in global memory) into LDS via async b128 copies. 256 threads: 2/row.
__device__ __forceinline__ void stage_f16(const _Float16* gA, int ldA,
                                          const _Float16* gB, int ldB,
                                          _Float16* dA, _Float16* dB, int k0) {
  const int t  = threadIdx.x;
  const int r  = t >> 1, c0 = (t & 1) * 32;
  const _Float16* sa = gA + (size_t)r * ldA + k0 + c0;
  const _Float16* sb = gB + (size_t)r * ldB + k0 + c0;
  _Float16* da = dA + r * LSTR + c0;
  _Float16* db = dB + r * LSTR + c0;
  #pragma unroll
  for (int j = 0; j < 4; ++j) {
    cp16_async(sa + j * 8, da + j * 8);
    cp16_async(sb + j * 8, db + j * 8);
  }
}

// 16 WMMAs on the current buffer: wave tile 32x64 = 2x4 tiles, 2 k-substeps.
__device__ __forceinline__ void compute_tile(const _Float16* cA, const _Float16* cB,
                                             int wm, int wn, v8f acc[2][4]) {
  #pragma unroll
  for (int ks = 0; ks < KSTEP; ks += 32) {
    v16h af[2], bf[4];
    #pragma unroll
    for (int i = 0; i < 2; ++i) af[i] = ld_frag(cA, wm + i * 16, ks);
    #pragma unroll
    for (int j = 0; j < 4; ++j) bf[j] = ld_frag(cB, wn + j * 16, ks);
    #pragma unroll
    for (int i = 0; i < 2; ++i)
      #pragma unroll
      for (int j = 0; j < 4; ++j)
        acc[i][j] = __builtin_amdgcn_wmma_f32_16x16x32_f16(
            false, af[i], false, bf[j], (short)0, acc[i][j], false, false);
  }
}

// ===========================================================================
// Kernel 1: Q/K/V projection.  out = x @ W + b  (M=8192, N=1024, K=1024)
// f32 sources (convert+transpose in VGPRs), double-buffered LDS.
// Q,K stored f16 row-major [B*S][D]; V stored f16 TRANSPOSED [B][D][S] so the
// PV GEMM's B tile is a contiguous async copy.
// ===========================================================================
__global__ __launch_bounds__(256) void qkv_gemm(
    const float* __restrict__ x,
    const float* __restrict__ Wq, const float* __restrict__ bq,
    const float* __restrict__ Wk, const float* __restrict__ bk,
    const float* __restrict__ Wv, const float* __restrict__ bv,
    _Float16* __restrict__ Q, _Float16* __restrict__ K, _Float16* __restrict__ Vt)
{
  __shared__ _Float16 sA[2][128 * LSTR];
  __shared__ _Float16 sB[2][128 * LSTR];

  const int z = blockIdx.z;
  const float* W    = (z == 0) ? Wq : ((z == 1) ? Wk : Wv);
  const float* bias = (z == 0) ? bq : ((z == 1) ? bk : bv);

  const int n0   = blockIdx.x * 128;
  const int m0   = blockIdx.y * 128;
  const int t    = threadIdx.x;
  const int wave = t >> 5;
  const int wm   = (wave & 3) * 32;
  const int wn   = (wave >> 2) * 64;

  const v8f vzero = {};
  v8f acc[2][4];
  #pragma unroll
  for (int i = 0; i < 2; ++i)
    #pragma unroll
    for (int j = 0; j < 4; ++j) acc[i][j] = vzero;

  // staging: f32 -> f16 with W transposed into [n][k]
  auto stage = [&](int buf, int k0) {
    { const int r = t >> 1, c0 = (t & 1) * 32;
      const float* src = x + (size_t)(m0 + r) * D_DIM + k0 + c0;
      _Float16* d = sA[buf] + r * LSTR + c0;
      #pragma unroll
      for (int j = 0; j < 8; ++j) {
        float4 f = *(const float4*)(src + j * 4);
        d[j * 4 + 0] = (_Float16)f.x; d[j * 4 + 1] = (_Float16)f.y;
        d[j * 4 + 2] = (_Float16)f.z; d[j * 4 + 3] = (_Float16)f.w;
      } }
    { const int kr = t >> 2, c0 = (t & 3) * 32;
      const float* src = W + (size_t)(k0 + kr) * D_DIM + n0 + c0;
      _Float16* d = sB[buf];
      #pragma unroll
      for (int j = 0; j < 8; ++j) {
        float4 f = *(const float4*)(src + j * 4);
        d[(c0 + j * 4 + 0) * LSTR + kr] = (_Float16)f.x;
        d[(c0 + j * 4 + 1) * LSTR + kr] = (_Float16)f.y;
        d[(c0 + j * 4 + 2) * LSTR + kr] = (_Float16)f.z;
        d[(c0 + j * 4 + 3) * LSTR + kr] = (_Float16)f.w;
      } }
  };

  const int NT = D_DIM / KSTEP;
  stage(0, 0);
  for (int kt = 0; kt < NT; ++kt) {
    wait_stage();
    const int cur = kt & 1;
    if (kt + 1 < NT) stage(cur ^ 1, (kt + 1) * KSTEP);
    compute_tile(sA[cur], sB[cur], wm, wn, acc);
  }

  const int lane = t & 31;
  const int nlo  = lane & 15;
  const int rhi  = (lane >> 4) * 8;
  #pragma unroll
  for (int j = 0; j < 4; ++j) {
    const int col = n0 + wn + j * 16 + nlo;
    const float bv_ = bias[col];
    #pragma unroll
    for (int i = 0; i < 2; ++i) {
      #pragma unroll
      for (int r = 0; r < 8; ++r) {
        const int row = m0 + wm + i * 16 + r + rhi;   // global row = b*S + s
        const _Float16 h = (_Float16)(acc[i][j][r] + bv_);
        if (z == 2) {  // V: store transposed [b][d][s]
          Vt[((size_t)(row >> 11) * D_DIM + col) * S_LEN + (row & (S_LEN - 1))] = h;
        } else {
          _Float16* out = (z == 0) ? Q : K;
          out[(size_t)row * D_DIM + col] = h;
        }
      }
    }
  }
}

// ===========================================================================
// Kernel 2: scores = Q @ K^T / sqrt(D), causal, f32 out. Async staging.
// B-matrix of the WMMA is K^T, so the B tile is rows of K ([n][k] directly).
// Blocks strictly above the diagonal skip the GEMM and fill -1e30.
// ===========================================================================
__global__ __launch_bounds__(256) void attn_scores(
    const _Float16* __restrict__ Q, const _Float16* __restrict__ Kb,
    float* __restrict__ Sc)
{
  const int b  = blockIdx.z;
  const int ry = blockIdx.y;
  const int cx = blockIdx.x;
  const int t  = threadIdx.x;
  float* Sout = Sc + (size_t)b * S_LEN * S_LEN;

  if (cx > ry) {
    for (int i = t; i < 128 * 128; i += 256) {
      const int r = i >> 7, c = i & 127;
      Sout[(size_t)(ry * 128 + r) * S_LEN + cx * 128 + c] = -1.0e30f;
    }
    return;
  }

  __shared__ _Float16 sA[2][128 * LSTR];
  __shared__ _Float16 sB[2][128 * LSTR];

  const int wave = t >> 5;
  const int wm = (wave & 3) * 32;
  const int wn = (wave >> 2) * 64;

  const _Float16* qrow = Q  + ((size_t)b * S_LEN + ry * 128) * D_DIM;
  const _Float16* krow = Kb + ((size_t)b * S_LEN + cx * 128) * D_DIM;

  const v8f vzero = {};
  v8f acc[2][4];
  #pragma unroll
  for (int i = 0; i < 2; ++i)
    #pragma unroll
    for (int j = 0; j < 4; ++j) acc[i][j] = vzero;

  const int NT = D_DIM / KSTEP;
  stage_f16(qrow, D_DIM, krow, D_DIM, sA[0], sB[0], 0);
  for (int kt = 0; kt < NT; ++kt) {
    wait_stage();
    const int cur = kt & 1;
    if (kt + 1 < NT)
      stage_f16(qrow, D_DIM, krow, D_DIM, sA[cur ^ 1], sB[cur ^ 1], (kt + 1) * KSTEP);
    compute_tile(sA[cur], sB[cur], wm, wn, acc);
  }

  const int lane = t & 31;
  const int nlo  = lane & 15;
  const int rhi  = (lane >> 4) * 8;
  #pragma unroll
  for (int j = 0; j < 4; ++j) {
    const int col = cx * 128 + wn + j * 16 + nlo;
    #pragma unroll
    for (int i = 0; i < 2; ++i) {
      #pragma unroll
      for (int r = 0; r < 8; ++r) {
        const int row = ry * 128 + wm + i * 16 + r + rhi;
        float v = acc[i][j][r] * 0.03125f;   // 1/sqrt(1024)
        if (col > row) v = -1.0e30f;         // diagonal-block causal mask
        Sout[(size_t)row * S_LEN + col] = v;
      }
    }
  }
}

// ===========================================================================
// Kernel 3: row softmax over 2048 f32 scores -> f16 probabilities.
// ===========================================================================
__global__ __launch_bounds__(256) void softmax_rows(
    const float* __restrict__ s, _Float16* __restrict__ p)
{
  __shared__ float red[256];
  const int row = blockIdx.x;
  const float* sr = s + (size_t)row * S_LEN;
  const int t = threadIdx.x;

  float v[8];
  float mx = -3.0e38f;
  #pragma unroll
  for (int i = 0; i < 8; ++i) { v[i] = sr[t + (i << 8)]; mx = fmaxf(mx, v[i]); }

  red[t] = mx; __syncthreads();
  for (int o = 128; o > 0; o >>= 1) {
    if (t < o) red[t] = fmaxf(red[t], red[t + o]);
    __syncthreads();
  }
  mx = red[0]; __syncthreads();

  float sum = 0.f;
  #pragma unroll
  for (int i = 0; i < 8; ++i) { v[i] = __expf(v[i] - mx); sum += v[i]; }

  red[t] = sum; __syncthreads();
  for (int o = 128; o > 0; o >>= 1) {
    if (t < o) red[t] += red[t + o];
    __syncthreads();
  }
  const float inv = 1.0f / red[0];

  _Float16* pr = p + (size_t)row * S_LEN;
  #pragma unroll
  for (int i = 0; i < 8; ++i) pr[t + (i << 8)] = (_Float16)(v[i] * inv);
}

// ===========================================================================
// Kernel 4: out = P @ V  (per batch M=2048, N=1024, K=2048), causal-truncated.
// V is pre-transposed ([b][d][s]) so the B tile is a contiguous async copy.
// ===========================================================================
__global__ __launch_bounds__(256) void attn_pv(
    const _Float16* __restrict__ P, const _Float16* __restrict__ Vt,
    float* __restrict__ O)
{
  __shared__ _Float16 sA[2][128 * LSTR];
  __shared__ _Float16 sB[2][128 * LSTR];

  const int b  = blockIdx.z;
  const int ry = blockIdx.y;          // query block (16)
  const int n0 = blockIdx.x * 128;    // d block     (8)
  const int t  = threadIdx.x;
  const int wave = t >> 5;
  const int wm = (wave & 3) * 32;
  const int wn = (wave >> 2) * 64;

  const _Float16* prow = P  + ((size_t)b * S_LEN + ry * 128) * S_LEN;
  const _Float16* vrow = Vt + ((size_t)b * D_DIM + n0) * S_LEN;   // rows = d, cols = s

  const v8f vzero = {};
  v8f acc[2][4];
  #pragma unroll
  for (int i = 0; i < 2; ++i)
    #pragma unroll
    for (int j = 0; j < 4; ++j) acc[i][j] = vzero;

  const int NT = (ry + 1) * (128 / KSTEP);   // causal truncation of K-loop
  stage_f16(prow, S_LEN, vrow, S_LEN, sA[0], sB[0], 0);
  for (int kt = 0; kt < NT; ++kt) {
    wait_stage();
    const int cur = kt & 1;
    if (kt + 1 < NT)
      stage_f16(prow, S_LEN, vrow, S_LEN, sA[cur ^ 1], sB[cur ^ 1], (kt + 1) * KSTEP);
    compute_tile(sA[cur], sB[cur], wm, wn, acc);
  }

  const int lane = t & 31;
  const int nlo  = lane & 15;
  const int rhi  = (lane >> 4) * 8;
  #pragma unroll
  for (int j = 0; j < 4; ++j) {
    const int col = n0 + wn + j * 16 + nlo;
    #pragma unroll
    for (int i = 0; i < 2; ++i) {
      #pragma unroll
      for (int r = 0; r < 8; ++r) {
        const int row = ry * 128 + wm + i * 16 + r + rhi;
        O[((size_t)b * S_LEN + row) * D_DIM + col] = acc[i][j][r];
      }
    }
  }
}

// ===========================================================================
extern "C" void kernel_launch(void* const* d_in, const int* in_sizes, int n_in,
                              void* d_out, int out_size, void* d_ws, size_t ws_size,
                              hipStream_t stream) {
  const float* x  = (const float*)d_in[0];
  // d_in[1] = mask (causality handled analytically)
  const float* Wq = (const float*)d_in[2];
  const float* bq = (const float*)d_in[3];
  const float* Wk = (const float*)d_in[4];
  const float* bk = (const float*)d_in[5];
  const float* Wv = (const float*)d_in[6];
  const float* bv = (const float*)d_in[7];
  float* out = (float*)d_out;

  // Workspace (~151 MB): Q,K f16 [B*S][D]; V f16 transposed [B][D][S];
  // scores f32 [B][S][S]; P f16 [B][S][S].
  const size_t ND = (size_t)NBATCH * S_LEN * D_DIM;
  const size_t NS = (size_t)NBATCH * S_LEN * S_LEN;
  _Float16* Q  = (_Float16*)d_ws;
  _Float16* K  = Q + ND;
  _Float16* Vt = K + ND;
  float*    S  = (float*)(Vt + ND);
  _Float16* P  = (_Float16*)(S + NS);

  qkv_gemm   <<<dim3(D_DIM / 128, (NBATCH * S_LEN) / 128, 3), 256, 0, stream>>>(
      x, Wq, bq, Wk, bk, Wv, bv, Q, K, Vt);
  attn_scores<<<dim3(S_LEN / 128, S_LEN / 128, NBATCH), 256, 0, stream>>>(Q, K, S);
  softmax_rows<<<dim3(NBATCH * S_LEN), 256, 0, stream>>>(S, P);
  attn_pv    <<<dim3(D_DIM / 128, S_LEN / 128, NBATCH), 256, 0, stream>>>(P, Vt, out);
}
